// NoGE_QGNN_DistMult_70437463654530
// MI455X (gfx1250) — matrix operand order repaired and checked by reading, not compiled, not run on hardware
//
#include <hip/hip_runtime.h>
#include <math.h>

typedef __attribute__((ext_vector_type(16))) _Float16 v16h;
typedef __attribute__((ext_vector_type(8)))  _Float16 v8h;
typedef __attribute__((ext_vector_type(8)))  float    v8f;

#define N_ENT 40000
#define N_REL 500
#define NTOT  40500
#define EMB   256
#define NEDGE 648000
#define NB    1024
#define MT1   2532   /* ceil(40500/16) */
#define NT2   2500   /* 40000/16 */

static __device__ __forceinline__ v16h hcat(v8h lo, v8h hi) {
  return __builtin_shufflevector(lo, hi, 0,1,2,3,4,5,6,7,8,9,10,11,12,13,14,15);
}

// Load one WMMA B fragment (two 8-half runs, 16 apart, of one column) per lane.
static __device__ __forceinline__ void loadBfrag(
    v16h* bf, const _Float16* __restrict__ base, int col, int koffB) {
  const _Float16* bcol = base + (size_t)col * EMB;
#pragma unroll
  for (int ks = 0; ks < 8; ++ks) {
    v8h lo = *(const v8h*)(bcol + ks * 32 + koffB);
    v8h hi = *(const v8h*)(bcol + ks * 32 + koffB + 16);
    bf[ks] = hcat(lo, hi);
  }
}

// ---------------------------------------------------------------- hamilton^T
__global__ __launch_bounds__(256) void k_build_hamT(
    const float* __restrict__ qgnn, _Float16* __restrict__ hamT) {
  const int k = threadIdx.x;        // row of hamilton
  const int n = blockIdx.x;         // col of hamilton
  const int a = k >> 6, b = k & 63;
  const int c = n >> 6, d = n & 63;
  const int   stab[4][4] = {{0,1,2,3},{1,0,3,2},{2,3,0,1},{3,2,1,0}};
  const float gtab[4][4] = {{1.f,-1.f,-1.f,-1.f},{1.f,1.f,-1.f,1.f},
                            {1.f,1.f,1.f,-1.f},{1.f,-1.f,1.f,1.f}};
  float v = gtab[c][a] * qgnn[b * EMB + stab[c][a] * 64 + d];
  hamT[n * EMB + k] = (_Float16)v;
}

// ---------------------------------------------------------------- X -> f16
__global__ __launch_bounds__(256) void k_embed_f16(
    const int* __restrict__ lst, const float* __restrict__ emb,
    _Float16* __restrict__ Xf16) {
  int t = blockIdx.x * 256 + threadIdx.x;        // NTOT*EMB threads
  int i = t >> 8, f = t & 255;
  Xf16[t] = (_Float16)emb[(size_t)lst[i] * EMB + f];
}

__global__ __launch_bounds__(256) void k_zero(float* __restrict__ p) {
  p[blockIdx.x * 256 + threadIdx.x] = 0.0f;
}

// ------------------------------------------------- GEMM1: support = X @ H
__global__ __launch_bounds__(256) void k_gemm_support(
    const _Float16* __restrict__ Xf16, const _Float16* __restrict__ hamT,
    float* __restrict__ support) {
  const int lane  = threadIdx.x & 31;
  const int wave  = threadIdx.x >> 5;
  const int mtile = blockIdx.x * 8 + wave;
  if (mtile >= MT1) return;                       // wave-uniform
  const int half = lane >> 4, l16 = lane & 15;
  const int koffA = half * 8, koffB = half * 8;
  const int row = mtile * 16 + l16;
  const bool fullTile = (mtile * 16 + 16) <= NTOT; // wave-uniform
  const _Float16* arow = Xf16 + (size_t)row * EMB;

  v16h a[8];
  if (fullTile) {
#pragma unroll
    for (int ks = 0; ks < 8; ++ks) {
      v8h lo = *(const v8h*)(arow + ks * 32 + koffA);
      v8h hi = *(const v8h*)(arow + ks * 32 + koffA + 16);
      a[ks] = hcat(lo, hi);
    }
  } else {
    const bool rowOK = row < NTOT;
#pragma unroll
    for (int ks = 0; ks < 8; ++ks) {
      v8h lo = {}, hi = {};
      if (rowOK) {
        lo = *(const v8h*)(arow + ks * 32 + koffA);
        hi = *(const v8h*)(arow + ks * 32 + koffA + 16);
      }
      a[ks] = hcat(lo, hi);
    }
  }

  v16h bf[2][8];
  loadBfrag(bf[0], hamT, 0 * 16 + l16, koffB);    // prologue prefetch
#pragma unroll
  for (int nt = 0; nt < 16; ++nt) {
    const int cur = nt & 1;
    if (nt < 15)                                   // prefetch next tile's B
      loadBfrag(bf[cur ^ 1], hamT, (nt + 1) * 16 + l16, koffB);
    __builtin_amdgcn_sched_barrier(0);             // keep prefetch above MMAs

    v8f acc = {};
#pragma unroll
    for (int ks = 0; ks < 8; ++ks)
      acc = __builtin_amdgcn_wmma_f32_16x16x32_f16(
          false, a[ks], false, bf[cur][ks], (short)0, acc, false, false);

    const int orow = mtile * 16 + half * 8;
    float* orp = support + (size_t)orow * EMB + nt * 16 + l16;
    if (fullTile) {
#pragma unroll
      for (int v = 0; v < 8; ++v) orp[(size_t)v * EMB] = acc[v];
    } else {
#pragma unroll
      for (int v = 0; v < 8; ++v)
        if (orow + v < NTOT) orp[(size_t)v * EMB] = acc[v];
    }
  }
}

// ------------------------------------------------- edge scatter (L2 atomics)
__global__ __launch_bounds__(256) void k_scatter(
    const int* __restrict__ esrc, const int* __restrict__ edst,
    const float* __restrict__ eval, const float* __restrict__ support,
    float* __restrict__ agg) {
  int t = blockIdx.x * 256 + threadIdx.x;
  int e = t >> 5, lane = t & 31;
  if (e >= NEDGE) return;
  int s = esrc[e], d = edst[e];
  float v = eval[e];
  const float4* sp = (const float4*)(support + (size_t)d * EMB + lane * 8);
  float4 p0 = sp[0], p1 = sp[1];
  float* ap = agg + (size_t)s * EMB + lane * 8;
  unsafeAtomicAdd(ap + 0, v * p0.x);
  unsafeAtomicAdd(ap + 1, v * p0.y);
  unsafeAtomicAdd(ap + 2, v * p0.z);
  unsafeAtomicAdd(ap + 3, v * p0.w);
  unsafeAtomicAdd(ap + 4, v * p1.x);
  unsafeAtomicAdd(ap + 5, v * p1.y);
  unsafeAtomicAdd(ap + 6, v * p1.z);
  unsafeAtomicAdd(ap + 7, v * p1.w);
}

// ------------------------------------------------- tanh (in place) + f16 copy
__global__ __launch_bounds__(256) void k_tanh(
    float* __restrict__ agg, _Float16* __restrict__ X2f16) {
  int t = blockIdx.x * 256 + threadIdx.x;
  float v = tanhf(agg[t]);
  agg[t] = v;
  X2f16[t] = (_Float16)v;
}

// ------------------------------------------------- hr = X2[e1] * X2[r+N_ENT]
__global__ __launch_bounds__(256) void k_hr(
    const int* __restrict__ e1, const int* __restrict__ ridx,
    const float* __restrict__ X2, float* __restrict__ hr) {
  int t = blockIdx.x * 256 + threadIdx.x;       // NB*EMB
  int b = t >> 8, f = t & 255;
  float h = X2[(size_t)e1[b] * EMB + f];
  float r = X2[(size_t)(ridx[b] + N_ENT) * EMB + f];
  hr[t] = h * r;
}

// ------------------------------------------------- BN batch stats (1 block)
__global__ __launch_bounds__(256) void k_bn_stats(
    const float* __restrict__ hr, const float* __restrict__ gamma,
    const float* __restrict__ beta, float* __restrict__ ss) {
  int f = threadIdx.x;
  float s = 0.f, s2 = 0.f;
  for (int b = 0; b < NB; ++b) {
    float x = hr[b * EMB + f];
    s += x; s2 += x * x;
  }
  float mean = s * (1.0f / NB);
  float var  = s2 * (1.0f / NB) - mean * mean;
  float sc   = gamma[f] * rsqrtf(var + 1e-5f);
  ss[f]       = sc;
  ss[EMB + f] = beta[f] - mean * sc;
}

__global__ __launch_bounds__(256) void k_bn_apply(
    const float* __restrict__ hr, const float* __restrict__ ss,
    _Float16* __restrict__ hrf16) {
  int t = blockIdx.x * 256 + threadIdx.x;
  int f = t & 255;
  hrf16[t] = (_Float16)(hr[t] * ss[f] + ss[EMB + f]);
}

// ------------------------------------------------- GEMM2: sigmoid(hr @ X2e^T)
__global__ __launch_bounds__(256) void k_gemm_score(
    const _Float16* __restrict__ hrf16, const _Float16* __restrict__ X2f16,
    float* __restrict__ out) {
  __shared__ _Float16 smem[16 * EMB];            // one 16-row hr strip (8 KB)
  const int mtile = blockIdx.y;
  {   // cooperative strip load: 512 uint4, 2 per thread
    const uint4* src = (const uint4*)(hrf16 + (size_t)mtile * 16 * EMB);
    uint4* dst = (uint4*)smem;
    dst[threadIdx.x]        = src[threadIdx.x];
    dst[threadIdx.x + 256]  = src[threadIdx.x + 256];
  }
  __syncthreads();

  const int lane = threadIdx.x & 31;
  const int wave = threadIdx.x >> 5;
  const int half = lane >> 4, l16 = lane & 15;
  const int koffA = half * 8, koffB = half * 8;

  v16h a[8];
  const _Float16* arow = smem + l16 * EMB;       // ds_load A fragments
#pragma unroll
  for (int ks = 0; ks < 8; ++ks) {
    v8h lo = *(const v8h*)(arow + ks * 32 + koffA);
    v8h hi = *(const v8h*)(arow + ks * 32 + koffA + 16);
    a[ks] = hcat(lo, hi);
  }

  const int nt0 = (blockIdx.x * 8 + wave) * 4;
  // NOTE: prefetching one tile past NT2 is memory-safe: max col touched is
  // 40447 < NTOT rows of X2f16. Only compute/store are range-guarded.
  v16h bf[2][8];
  loadBfrag(bf[0], X2f16, nt0 * 16 + l16, koffB);
#pragma unroll
  for (int i = 0; i < 4; ++i) {
    const int cur = i & 1;
    if (i < 3)
      loadBfrag(bf[cur ^ 1], X2f16, (nt0 + i + 1) * 16 + l16, koffB);
    __builtin_amdgcn_sched_barrier(0);

    const int ntile = nt0 + i;
    if (ntile < NT2) {                            // wave-uniform guard
      v8f acc = {};
#pragma unroll
      for (int ks = 0; ks < 8; ++ks)
        acc = __builtin_amdgcn_wmma_f32_16x16x32_f16(
            false, a[ks], false, bf[cur][ks], (short)0, acc, false, false);

      const int orow = mtile * 16 + half * 8;
      const int col  = ntile * 16 + l16;
#pragma unroll
      for (int v = 0; v < 8; ++v) {
        float sgm = 1.0f / (1.0f + expf(-acc[v]));
        // streaming output (164 MB, never re-read): keep it out of L2
        __builtin_nontemporal_store(sgm, out + (size_t)(orow + v) * N_ENT + col);
      }
    }
  }
}

// =====================================================================
extern "C" void kernel_launch(void* const* d_in, const int* in_sizes, int n_in,
                              void* d_out, int out_size, void* d_ws, size_t ws_size,
                              hipStream_t stream) {
  const int*   e1_idx  = (const int*)d_in[0];
  const int*   r_idx   = (const int*)d_in[1];
  const int*   lst     = (const int*)d_in[2];
  const float* emb     = (const float*)d_in[3];
  const float* qgnn    = (const float*)d_in[4];
  const int*   esrc    = (const int*)d_in[5];
  const int*   edst    = (const int*)d_in[6];
  const float* eval    = (const float*)d_in[7];
  const float* gamma   = (const float*)d_in[8];
  const float* beta    = (const float*)d_in[9];
  float* out = (float*)d_out;

  char* ws = (char*)d_ws;
  const size_t szXf16 = (size_t)NTOT * EMB * 2;   // 20,736,000
  const size_t szHam  = (size_t)EMB * EMB * 2;    // 131,072
  const size_t szF32  = (size_t)NTOT * EMB * 4;   // 41,472,000
  const size_t szHr   = (size_t)NB * EMB * 4;
  _Float16* Xf16    = (_Float16*)(ws);
  _Float16* hamT    = (_Float16*)(ws + szXf16);
  float*    support = (float*)   (ws + szXf16 + szHam);
  float*    agg     = (float*)   (ws + szXf16 + szHam + szF32);
  _Float16* X2f16   = (_Float16*)(ws + szXf16 + szHam + 2 * szF32);
  float*    hr      = (float*)   (ws + 2 * szXf16 + szHam + 2 * szF32);
  float*    ss      = (float*)   (ws + 2 * szXf16 + szHam + 2 * szF32 + szHr);
  _Float16* hrf16   = (_Float16*)(ws + 2 * szXf16 + szHam + 2 * szF32 + szHr + 2048);

  k_build_hamT <<<EMB, 256, 0, stream>>>(qgnn, hamT);
  k_embed_f16  <<<NTOT, 256, 0, stream>>>(lst, emb, Xf16);
  k_zero       <<<NTOT, 256, 0, stream>>>(agg);
  k_gemm_support<<<(MT1 + 7) / 8, 256, 0, stream>>>(Xf16, hamT, support);
  k_scatter    <<<(NEDGE * 32) / 256, 256, 0, stream>>>(esrc, edst, eval, support, agg);
  k_tanh       <<<NTOT, 256, 0, stream>>>(agg, X2f16);
  k_hr         <<<NB, 256, 0, stream>>>(e1_idx, r_idx, agg, hr);
  k_bn_stats   <<<1, 256, 0, stream>>>(hr, gamma, beta, ss);
  k_bn_apply   <<<NB, 256, 0, stream>>>(hr, ss, hrf16);
  k_gemm_score <<<dim3(79, 64), 256, 0, stream>>>(hrf16, X2f16, out);
}
